// TtMoeLayer_45414984188606
// MI455X (gfx1250) — matrix operand order repaired and checked by reading, not compiled, not run on hardware
//
#include <hip/hip_runtime.h>
#include <hip/hip_bf16.h>
#include <stdint.h>

#define NTOK   8192
#define HDIM   4096
#define ODIM   4096
#define NEXP   8
#define MTILE  128
#define NTILE  128
#define MAXMT  72                 // max M-tiles per pass: 8192/128 + 7 = 71
#define PERMCAP (MAXMT * MTILE)   // 9216
#define NCHUNK (HDIM / 64)

#ifndef HAVE_TDM
#if defined(__has_builtin)
#if __has_builtin(__builtin_amdgcn_tensor_load_to_lds)
#define HAVE_TDM 1
#endif
#endif
#endif
#ifndef HAVE_TDM
#define HAVE_TDM 0
#endif

typedef __attribute__((ext_vector_type(16))) __bf16 v16bf;
typedef __attribute__((ext_vector_type(8)))  float  v8f;
typedef unsigned int v4u __attribute__((ext_vector_type(4)));
typedef int          v8i __attribute__((ext_vector_type(8)));
typedef int          v4i __attribute__((ext_vector_type(4)));

union Frag16 { v16bf v; uint4 q[2]; };

static __device__ __forceinline__ unsigned short f2bf(float f) {
    union { float f; unsigned u; } c; c.f = f;
    unsigned u = c.u + 0x7FFFu + ((c.u >> 16) & 1u);   // round-to-nearest-even
    return (unsigned short)(u >> 16);
}

#if HAVE_TDM
// TDM: load a 128(row) x 64(bf16) tile from a row-major [4096][4096] bf16 tensor
// into LDS, padding each 128B row with 16B so LDS row stride = 144B (72 bf16).
// D# per CDNA5 ISA ch.8: group0 = {count|flags, lds_addr, gaddr_lo, gaddr_hi|type},
// group1 packs data_size/pad/tensor dims/tile dims/stride0.
static __device__ __forceinline__ void tdm_load_b(unsigned lds_addr,
                                                  unsigned long long gaddr) {
    v4u g0;
    g0.x = 1u;                                            // count=1, user desc
    g0.y = lds_addr;                                      // LDS byte address
    g0.z = (unsigned)gaddr;                               // global_addr[31:0]
    g0.w = ((unsigned)(gaddr >> 32) & 0x1FFFFFFu) | 0x80000000u; // [56:32] | type=2
    v8i g1;
    g1[0] = 0x07110000;  // wg_mask=0, data_size=1(2B), pad_en=1, pad_int=4(128B), pad_amt=3(16B)
    g1[1] = 0x10000000;  // atomic_barrier_addr=0, tensor_dim0[15:0]=4096
    g1[2] = 0x10000000;  // tensor_dim0[31:16]=0, tensor_dim1[15:0]=4096
    g1[3] = 0x00400000;  // tensor_dim1[31:16]=0, tile_dim0=64
    g1[4] = 0x00000080;  // tile_dim1=128, tile_dim2=0
    g1[5] = 4096;        // tensor_dim0_stride[31:0]
    g1[6] = 0;           // stride0[47:32]=0, stride1[15:0]=0
    g1[7] = 0;           // stride1[47:16]=0
    v4i gz = {0, 0, 0, 0};
#if defined(__clang_major__) && (__clang_major__ >= 23)
    v8i gz8 = {0, 0, 0, 0, 0, 0, 0, 0};
    __builtin_amdgcn_tensor_load_to_lds(g0, g1, gz, gz, gz8, 0);
#else
    __builtin_amdgcn_tensor_load_to_lds(g0, g1, gz, gz, 0);
#endif
}
#endif

// ---------------- kernel: zero the tiny counters ----------------
__global__ void moe_init(int* __restrict__ counts, int* __restrict__ fill) {
    if (threadIdx.x < 16) { counts[threadIdx.x] = 0; fill[threadIdx.x] = 0; }
}

// ---------------- kernel: gating (one wave per token) ----------------
__global__ __launch_bounds__(256)
void moe_gate(const float* __restrict__ x, const float* __restrict__ gw,
              int* __restrict__ tok_e, float* __restrict__ tok_w,
              int* __restrict__ counts) {
    int wave = threadIdx.x >> 5, lane = threadIdx.x & 31;
    int t = blockIdx.x * 8 + wave;
    const float* xr = x + (size_t)t * HDIM;
    float acc[NEXP];
#pragma unroll
    for (int e = 0; e < NEXP; ++e) acc[e] = 0.f;
    for (int h = lane; h < HDIM; h += 32) {
        float xv = xr[h];
        const float* g = gw + h * NEXP;
#pragma unroll
        for (int e = 0; e < NEXP; ++e) acc[e] += xv * g[e];
    }
#pragma unroll
    for (int e = 0; e < NEXP; ++e)
        for (int off = 16; off > 0; off >>= 1)
            acc[e] += __shfl_down(acc[e], off);
    if (lane == 0) {
        int e0 = 0; float v0 = acc[0];
#pragma unroll
        for (int e = 1; e < NEXP; ++e) if (acc[e] > v0) { v0 = acc[e]; e0 = e; }
        int e1 = (e0 == 0) ? 1 : 0; float v1 = -3.0e38f;
#pragma unroll
        for (int e = 0; e < NEXP; ++e)
            if (e != e0 && acc[e] > v1) { v1 = acc[e]; e1 = e; }
        float b = __expf(v1 - v0);        // v0 >= v1
        float s = 1.0f + b;
        tok_e[2 * t]     = e0;  tok_e[2 * t + 1] = e1;
        tok_w[2 * t]     = 1.0f / s;
        tok_w[2 * t + 1] = b / s;
        atomicAdd(&counts[e0], 1);
        atomicAdd(&counts[8 + e1], 1);
    }
}

// ---------------- kernel: segment offsets + tile->expert map ----------------
__global__ void moe_setup(const int* __restrict__ counts, int* __restrict__ offs,
                          int* __restrict__ tileExp, int* __restrict__ perm) {
    for (int i = threadIdx.x; i < 2 * PERMCAP; i += blockDim.x) perm[i] = -1;
    if (threadIdx.x == 0) {
        for (int p = 0; p < 2; ++p) {
            int tp = 0;
            for (int e = 0; e < NEXP; ++e) {
                offs[p * 8 + e] = tp * MTILE;
                int tiles = (counts[p * 8 + e] + MTILE - 1) / MTILE;
                for (int k = 0; k < tiles; ++k) tileExp[p * MAXMT + tp + k] = e;
                tp += tiles;
            }
            for (; tp < MAXMT; ++tp) tileExp[p * MAXMT + tp] = -1;
        }
    }
}

// ---------------- kernel: scatter tokens into permuted slots ----------------
__global__ void moe_scatter(const int* __restrict__ tok_e, const int* __restrict__ offs,
                            int* __restrict__ fill, int* __restrict__ perm) {
    int t = blockIdx.x * blockDim.x + threadIdx.x;
    if (t >= NTOK) return;
#pragma unroll
    for (int p = 0; p < 2; ++p) {
        int e = tok_e[2 * t + p];
        int idx = atomicAdd(&fill[p * 8 + e], 1);
        perm[p * PERMCAP + offs[p * 8 + e] + idx] = t;
    }
}

// ---------------- kernel: x fp32 -> bf16 ----------------
__global__ __launch_bounds__(256)
void conv_x(const float* __restrict__ x, unsigned short* __restrict__ xq) {
    size_t i = ((size_t)blockIdx.x * blockDim.x + threadIdx.x) * 8;
    const float4* p = (const float4*)(x + i);
    float4 f0 = p[0], f1 = p[1];
    float v[8] = {f0.x, f0.y, f0.z, f0.w, f1.x, f1.y, f1.z, f1.w};
    unsigned q[4];
#pragma unroll
    for (int j = 0; j < 4; ++j)
        q[j] = (unsigned)f2bf(v[2 * j]) | ((unsigned)f2bf(v[2 * j + 1]) << 16);
    *(uint4*)(xq + i) = make_uint4(q[0], q[1], q[2], q[3]);
}

// ---------------- kernel: expert_w fp32 [e][k][n] -> bf16 transposed [e][n][k] ----------------
__global__ __launch_bounds__(256)
void conv_wt(const float* __restrict__ w, unsigned short* __restrict__ wt) {
    int e  = blockIdx.z;
    int k0 = blockIdx.y * 64;
    int n0 = blockIdx.x * 64;
    const float* W = w + (size_t)e * HDIM * ODIM;
    unsigned short* WT = wt + (size_t)e * HDIM * ODIM;
    __shared__ __align__(16) unsigned short T[64][72];

    int t = threadIdx.x;
    {
        int kr = t >> 2, nc = (t & 3) * 16;
        const float4* p = (const float4*)(W + (size_t)(k0 + kr) * ODIM + n0 + nc);
        float4 f0 = p[0], f1 = p[1], f2 = p[2], f3 = p[3];
        float v[16] = {f0.x, f0.y, f0.z, f0.w, f1.x, f1.y, f1.z, f1.w,
                       f2.x, f2.y, f2.z, f2.w, f3.x, f3.y, f3.z, f3.w};
        unsigned q[8];
#pragma unroll
        for (int j = 0; j < 8; ++j)
            q[j] = (unsigned)f2bf(v[2 * j]) | ((unsigned)f2bf(v[2 * j + 1]) << 16);
        *(uint4*)&T[kr][nc]     = make_uint4(q[0], q[1], q[2], q[3]);
        *(uint4*)&T[kr][nc + 8] = make_uint4(q[4], q[5], q[6], q[7]);
    }
    __syncthreads();
    {
        int nr = t >> 2, kc = (t & 3) * 16;
        unsigned q[8];
#pragma unroll
        for (int j = 0; j < 8; ++j)
            q[j] = (unsigned)T[kc + 2 * j][nr] | ((unsigned)T[kc + 2 * j + 1][nr] << 16);
        uint4* o = (uint4*)(WT + (size_t)(n0 + nr) * HDIM + k0 + kc);
        o[0] = make_uint4(q[0], q[1], q[2], q[3]);
        o[1] = make_uint4(q[4], q[5], q[6], q[7]);
    }
}

// ======== fast path: grouped GEMM, bf16 WMMA; B tile staged by TDM ========
__global__ __launch_bounds__(256)
void moe_gemm_bf16(const unsigned short* __restrict__ xq,
                   const unsigned short* __restrict__ wt,
                   const float* __restrict__ tok_w, const int* __restrict__ perm,
                   const int* __restrict__ tileExp, float* __restrict__ out, int pass) {
    int mt = blockIdx.y, nt = blockIdx.x;
    int ex = tileExp[pass * MAXMT + mt];
    if (ex < 0) return;
    const int* permP = perm + pass * PERMCAP + mt * MTILE;
    const unsigned short* W = wt + (size_t)ex * HDIM * ODIM;   // [n][k] bf16

    __shared__ __align__(16) unsigned short As[MTILE][72];
#if HAVE_TDM
    __shared__ __align__(16) unsigned short Bs[2][NTILE][72];
#else
    __shared__ __align__(16) unsigned short Bs[1][NTILE][72];
#endif

    int tid = threadIdx.x;
    int ar = tid >> 1, akh = (tid & 1) * 32;
    int tokA = permP[ar];
    const unsigned short* aptr = xq + (size_t)((tokA >= 0) ? tokA : 0) * HDIM + akh;

    int wave = tid >> 5, lane = tid & 31;
    int wm = (wave & 1) * 64;
    int wn = (wave >> 1) * 32;
    int lm = lane & 15, lh = lane >> 4;

#if HAVE_TDM
    unsigned long long gB = (unsigned long long)(uintptr_t)(W + (size_t)(nt * NTILE) * HDIM);
    unsigned ldsB0 = (unsigned)(uintptr_t)&Bs[0][0][0];
    unsigned ldsB1 = (unsigned)(uintptr_t)&Bs[1][0][0];
    if (wave == 0) tdm_load_b(ldsB0, gB);          // prefetch chunk 0
#else
    int bn = tid >> 1, bkh = (tid & 1) * 32;
    const unsigned short* bptr = W + (size_t)(nt * NTILE + bn) * HDIM + bkh;
#endif

    v8f acc[4][2];
#pragma unroll
    for (int i = 0; i < 4; ++i)
#pragma unroll
        for (int j = 0; j < 2; ++j) { v8f z = {}; acc[i][j] = z; }

    for (int i = 0; i < NCHUNK; ++i) {
        int kk = i * 64;
        uint4 a0{}, a1{}, a2{}, a3{};
        if (tokA >= 0) {
            const uint4* pa = (const uint4*)(aptr + kk);
            a0 = pa[0]; a1 = pa[1]; a2 = pa[2]; a3 = pa[3];
        }
#if !HAVE_TDM
        const uint4* pb = (const uint4*)(bptr + kk);
        uint4 b0 = pb[0], b1 = pb[1], b2 = pb[2], b3 = pb[3];
#endif

        __syncthreads();    // prior chunk's compute done: As free, other B buf free
#if HAVE_TDM
        if (wave == 0) {
            if (i + 1 < NCHUNK)
                tdm_load_b(((i + 1) & 1) ? ldsB1 : ldsB0, gB + (size_t)(kk + 64) * 2);
        }
#endif
        *(uint4*)&As[ar][akh]      = a0;
        *(uint4*)&As[ar][akh + 8]  = a1;
        *(uint4*)&As[ar][akh + 16] = a2;
        *(uint4*)&As[ar][akh + 24] = a3;
#if !HAVE_TDM
        *(uint4*)&Bs[0][bn][bkh]      = b0;
        *(uint4*)&Bs[0][bn][bkh + 8]  = b1;
        *(uint4*)&Bs[0][bn][bkh + 16] = b2;
        *(uint4*)&Bs[0][bn][bkh + 24] = b3;
#endif
#if HAVE_TDM
        if (wave == 0) {
            if (i + 1 < NCHUNK) __builtin_amdgcn_s_wait_tensorcnt(1); // chunk i arrived
            else                __builtin_amdgcn_s_wait_tensorcnt(0);
        }
#endif
        __syncthreads();    // As visible + B chunk i published

#if HAVE_TDM
        const unsigned short (*Bcur)[72] = Bs[i & 1];
#else
        const unsigned short (*Bcur)[72] = Bs[0];
#endif
#pragma unroll
        for (int ks = 0; ks < 64; ks += 32) {
            Frag16 afr[4], bfr[2];
#pragma unroll
            for (int mf = 0; mf < 4; ++mf) {
                int m = wm + mf * 16 + lm;
                afr[mf].q[0] = *(const uint4*)&As[m][ks + lh * 8];
                afr[mf].q[1] = *(const uint4*)&As[m][ks + 16 + lh * 8];
            }
#pragma unroll
            for (int nf = 0; nf < 2; ++nf) {
                int n = wn + nf * 16 + lm;
                bfr[nf].q[0] = *(const uint4*)&Bcur[n][ks + lh * 8];
                bfr[nf].q[1] = *(const uint4*)&Bcur[n][ks + 16 + lh * 8];
            }
#pragma unroll
            for (int mf = 0; mf < 4; ++mf)
#pragma unroll
                for (int nf = 0; nf < 2; ++nf)
                    acc[mf][nf] = __builtin_amdgcn_wmma_f32_16x16x32_bf16(
                        false, afr[mf].v, false, bfr[nf].v,
                        (short)0, acc[mf][nf], false, false);
        }
    }

    int colBase = nt * NTILE + wn + lm;
#pragma unroll
    for (int mf = 0; mf < 4; ++mf) {
#pragma unroll
        for (int r = 0; r < 8; ++r) {
            int rowLocal = wm + mf * 16 + lh * 8 + r;
            int token = permP[rowLocal];
            if (token < 0) continue;
            float wgt = tok_w[2 * token + pass];
            float* o = out + (size_t)token * ODIM + colBase;
#pragma unroll
            for (int nf = 0; nf < 2; ++nf) {
                float val = wgt * acc[mf][nf][r];
                if (pass == 0) o[nf * 16] = val;
                else           o[nf * 16] += val;
            }
        }
    }
}

// ======== fallback: grouped GEMM with in-loop fp32->bf16 (small ws) ========
__global__ __launch_bounds__(256)
void moe_gemm_cvt(const float* __restrict__ x, const float* __restrict__ ew,
                  const float* __restrict__ tok_w, const int* __restrict__ perm,
                  const int* __restrict__ tileExp, float* __restrict__ out, int pass) {
    int mt = blockIdx.y, nt = blockIdx.x;
    int ex = tileExp[pass * MAXMT + mt];
    if (ex < 0) return;
    const int* permP = perm + pass * PERMCAP + mt * MTILE;
    const float* W = ew + (size_t)ex * HDIM * ODIM;

    __shared__ __align__(16) unsigned short As[MTILE][40];
    __shared__ __align__(16) unsigned short Bs[NTILE][40];

    int tid = threadIdx.x;
    int ar  = tid >> 1;
    int akh = (tid & 1) * 16;
    int tokA = permP[ar];
    const float* aptr = (tokA >= 0) ? (x + (size_t)tokA * HDIM + akh) : x;
    int bk = (tid >> 4) * 2;
    int bn = (tid & 15) * 8;
    const float* bptr = W + (size_t)bk * ODIM + (size_t)nt * NTILE + bn;

    int wave = tid >> 5, lane = tid & 31;
    int wm = (wave & 1) * 64;
    int wn = (wave >> 1) * 32;
    int lm = lane & 15, lh = lane >> 4;

    v8f acc[4][2];
#pragma unroll
    for (int i = 0; i < 4; ++i)
#pragma unroll
        for (int j = 0; j < 2; ++j) { v8f z = {}; acc[i][j] = z; }

    for (int kk = 0; kk < HDIM; kk += 32) {
        float4 a0{}, a1{}, a2{}, a3{};
        if (tokA >= 0) {
            const float4* p = (const float4*)(aptr + kk);
            a0 = p[0]; a1 = p[1]; a2 = p[2]; a3 = p[3];
        }
        const float4* q0 = (const float4*)(bptr + (size_t)kk * ODIM);
        float4 b00 = q0[0], b01 = q0[1];
        const float4* q1 = (const float4*)(bptr + (size_t)(kk + 1) * ODIM);
        float4 b10 = q1[0], b11 = q1[1];

        __syncthreads();
        {
            float av[16] = {a0.x, a0.y, a0.z, a0.w, a1.x, a1.y, a1.z, a1.w,
                            a2.x, a2.y, a2.z, a2.w, a3.x, a3.y, a3.z, a3.w};
            unsigned ap[8];
#pragma unroll
            for (int j = 0; j < 8; ++j)
                ap[j] = (unsigned)f2bf(av[2 * j]) | ((unsigned)f2bf(av[2 * j + 1]) << 16);
            *(uint4*)&As[ar][akh]     = make_uint4(ap[0], ap[1], ap[2], ap[3]);
            *(uint4*)&As[ar][akh + 8] = make_uint4(ap[4], ap[5], ap[6], ap[7]);
        }
        {
            float r0[8] = {b00.x, b00.y, b00.z, b00.w, b01.x, b01.y, b01.z, b01.w};
            float r1[8] = {b10.x, b10.y, b10.z, b10.w, b11.x, b11.y, b11.z, b11.w};
#pragma unroll
            for (int i = 0; i < 8; ++i) {
                unsigned v = (unsigned)f2bf(r0[i]) | ((unsigned)f2bf(r1[i]) << 16);
                *(unsigned*)&Bs[bn + i][bk] = v;
            }
        }
        __syncthreads();

        Frag16 afr[4], bfr[2];
#pragma unroll
        for (int mf = 0; mf < 4; ++mf) {
            int m = wm + mf * 16 + lm;
            afr[mf].q[0] = *(const uint4*)&As[m][lh * 8];
            afr[mf].q[1] = *(const uint4*)&As[m][16 + lh * 8];
        }
#pragma unroll
        for (int nf = 0; nf < 2; ++nf) {
            int n = wn + nf * 16 + lm;
            bfr[nf].q[0] = *(const uint4*)&Bs[n][lh * 8];
            bfr[nf].q[1] = *(const uint4*)&Bs[n][16 + lh * 8];
        }
#pragma unroll
        for (int mf = 0; mf < 4; ++mf)
#pragma unroll
            for (int nf = 0; nf < 2; ++nf)
                acc[mf][nf] = __builtin_amdgcn_wmma_f32_16x16x32_bf16(
                    false, afr[mf].v, false, bfr[nf].v,
                    (short)0, acc[mf][nf], false, false);
    }

    int colBase = nt * NTILE + wn + lm;
#pragma unroll
    for (int mf = 0; mf < 4; ++mf) {
#pragma unroll
        for (int r = 0; r < 8; ++r) {
            int rowLocal = wm + mf * 16 + lh * 8 + r;
            int token = permP[rowLocal];
            if (token < 0) continue;
            float wgt = tok_w[2 * token + pass];
            float* o = out + (size_t)token * ODIM + colBase;
#pragma unroll
            for (int nf = 0; nf < 2; ++nf) {
                float val = wgt * acc[mf][nf][r];
                if (pass == 0) o[nf * 16] = val;
                else           o[nf * 16] += val;
            }
        }
    }
}

extern "C" void kernel_launch(void* const* d_in, const int* in_sizes, int n_in,
                              void* d_out, int out_size, void* d_ws, size_t ws_size,
                              hipStream_t stream) {
    const float* x  = (const float*)d_in[0];   // [8192, 1, 4096]
    const float* gw = (const float*)d_in[1];   // [4096, 8]
    const float* ew = (const float*)d_in[2];   // [8, 4096, 4096]
    float* out = (float*)d_out;                // [8192, 1, 4096]

    const size_t xqElems = (size_t)NTOK * HDIM;
    const size_t wtElems = (size_t)NEXP * HDIM * ODIM;
    const size_t ctrlInts = 2 * NTOK + 2 * NTOK + 16 + 16 + 16 + 2 * MAXMT + 2 * PERMCAP;
    const size_t bigNeed = (xqElems + wtElems) * sizeof(unsigned short)
                         + ctrlInts * sizeof(int);

    dim3 grid(ODIM / NTILE, MAXMT);

    if (ws_size >= bigNeed) {
        unsigned short* xq  = (unsigned short*)d_ws;
        unsigned short* wtq = xq + xqElems;
        int*   tok_e   = (int*)(wtq + wtElems);
        float* tok_w   = (float*)(tok_e + 2 * NTOK);
        int*   counts  = (int*)(tok_w + 2 * NTOK);
        int*   offs    = counts + 16;
        int*   fill    = offs + 16;
        int*   tileExp = fill + 16;
        int*   perm    = tileExp + 2 * MAXMT;

        conv_x <<<(int)(xqElems / (256 * 8)), 256, 0, stream>>>(x, xq);
        conv_wt<<<dim3(ODIM / 64, HDIM / 64, NEXP), 256, 0, stream>>>(ew, wtq);

        moe_init   <<<1, 64, 0, stream>>>(counts, fill);
        moe_gate   <<<NTOK / 8, 256, 0, stream>>>(x, gw, tok_e, tok_w, counts);
        moe_setup  <<<1, 256, 0, stream>>>(counts, offs, tileExp, perm);
        moe_scatter<<<NTOK / 256, 256, 0, stream>>>(tok_e, offs, fill, perm);

        moe_gemm_bf16<<<grid, 256, 0, stream>>>(xq, wtq, tok_w, perm, tileExp, out, 0);
        moe_gemm_bf16<<<grid, 256, 0, stream>>>(xq, wtq, tok_w, perm, tileExp, out, 1);
    } else {
        int*   tok_e   = (int*)d_ws;
        float* tok_w   = (float*)(tok_e + 2 * NTOK);
        int*   counts  = (int*)(tok_w + 2 * NTOK);
        int*   offs    = counts + 16;
        int*   fill    = offs + 16;
        int*   tileExp = fill + 16;
        int*   perm    = tileExp + 2 * MAXMT;

        moe_init   <<<1, 64, 0, stream>>>(counts, fill);
        moe_gate   <<<NTOK / 8, 256, 0, stream>>>(x, gw, tok_e, tok_w, counts);
        moe_setup  <<<1, 256, 0, stream>>>(counts, offs, tileExp, perm);
        moe_scatter<<<NTOK / 256, 256, 0, stream>>>(tok_e, offs, fill, perm);

        moe_gemm_cvt<<<grid, 256, 0, stream>>>(x, ew, tok_w, perm, tileExp, out, 0);
        moe_gemm_cvt<<<grid, 256, 0, stream>>>(x, ew, tok_w, perm, tileExp, out, 1);
    }
}
